// HTALayer_72911364817309
// MI455X (gfx1250) — compile-verified, matrix operand order checked by reading, not run on hardware
//
#include <hip/hip_runtime.h>
#include <hip/hip_bf16.h>
#include <math.h>

// Problem constants (from the reference)
#define NN     2048
#define MM     512
#define DD     256
#define NHEADS 8
#define HDIM   32
#define RANK_  32

typedef __attribute__((ext_vector_type(16))) _Float16 v16h;
typedef __attribute__((ext_vector_type(8)))  _Float16 v8h;
typedef __attribute__((ext_vector_type(8)))  float    v8f;

static __device__ __forceinline__ v8f wmma16x16x32(v16h a, v16h b, v8f c) {
  // D = A(16x32 f16) * B(32x16 f16) + C(16x16 f32)
  return __builtin_amdgcn_wmma_f32_16x16x32_f16(false, a, false, b, (short)0, c,
                                                false, false);
}

// ---------------------------------------------------------------------------
// Pack a row-major f32 matrix (R x K, leading dim lda; optionally read
// transposed: element[row][k] = src[k*lda + row]) into WMMA operand tiles:
//   [R/16 row-tiles][K/32 k-tiles][32 lanes][16 halves]
// Lane half lh owns K interleave: h<8 -> K=(h&7)+8*lh ; h>=8 -> K=16+(h&7)+8*lh
// Both A and B^T operands of C = A*B^T pack identically with this layout.
// ---------------------------------------------------------------------------
__global__ __launch_bounds__(256) void pack_f16_kernel(
    const float* __restrict__ src, _Float16* __restrict__ dst,
    int R, int K, int lda, int trans)
{
  int idx = blockIdx.x * 256 + threadIdx.x;
  if (idx >= R * K) return;
  int h      = idx & 15;
  int lane   = (idx >> 4) & 31;
  int tile   = idx >> 9;
  int ktiles = K >> 5;
  int kt = tile % ktiles;
  int rt = tile / ktiles;
  int lh = lane >> 4;
  int lm = lane & 15;
  int row = rt * 16 + lm;
  int k   = kt * 32 + (h & 7) + 8 * (((h >> 3) << 1) + lh);
  float v = trans ? src[(size_t)k * lda + row] : src[(size_t)row * lda + k];
  dst[idx] = (_Float16)v;
}

// ---------------------------------------------------------------------------
// Packed WMMA GEMM: C[M,N] = A[M,K] * B[N,K]^T, f32 output.
// 16x64 output strip per wave (4 accumulators -> A-operand reused 4x), with
// explicit software pipelining: operands for k+1 are loaded before the k-step
// WMMAs so the loads stay in flight behind the matrix pipe (partial waits
// instead of wait-all before each WMMA).
// tilesN must be a multiple of 4 (true for all call sites: N = 256 or 2048).
// ---------------------------------------------------------------------------
__global__ __launch_bounds__(256) void gemm_pk_kernel(
    const _Float16* __restrict__ Apk, const _Float16* __restrict__ Bpk,
    float* __restrict__ C, int tilesM, int tilesN, int kt, int ldc)
{
  int wave = (blockIdx.x * 256 + threadIdx.x) >> 5;
  int lane = threadIdx.x & 31;
  int tn4 = tilesN >> 2;
  if (wave >= tilesM * tn4) return;               // uniform per wave
  int tm = wave / tn4;
  int tn = (wave % tn4) << 2;
  const _Float16* ap = Apk + ((size_t)tm * kt) * 512 + lane * 16;
  const _Float16* bp = Bpk + ((size_t)tn * kt) * 512 + lane * 16;
  size_t bs = (size_t)kt * 512;
  v8f acc0 = {}, acc1 = {}, acc2 = {}, acc3 = {};

  v16h va  = *(const v16h*)(ap);
  v16h vb0 = *(const v16h*)(bp);
  v16h vb1 = *(const v16h*)(bp + bs);
  v16h vb2 = *(const v16h*)(bp + 2 * bs);
  v16h vb3 = *(const v16h*)(bp + 3 * bs);
  for (int k = 0; k + 1 < kt; ++k) {
    size_t o = (size_t)(k + 1) * 512;
    v16h na  = *(const v16h*)(ap + o);
    v16h nb0 = *(const v16h*)(bp + o);
    v16h nb1 = *(const v16h*)(bp + o + bs);
    v16h nb2 = *(const v16h*)(bp + o + 2 * bs);
    v16h nb3 = *(const v16h*)(bp + o + 3 * bs);
    acc0 = wmma16x16x32(va, vb0, acc0);
    acc1 = wmma16x16x32(va, vb1, acc1);
    acc2 = wmma16x16x32(va, vb2, acc2);
    acc3 = wmma16x16x32(va, vb3, acc3);
    va = na; vb0 = nb0; vb1 = nb1; vb2 = nb2; vb3 = nb3;
  }
  acc0 = wmma16x16x32(va, vb0, acc0);
  acc1 = wmma16x16x32(va, vb1, acc1);
  acc2 = wmma16x16x32(va, vb2, acc2);
  acc3 = wmma16x16x32(va, vb3, acc3);

  int lh = lane >> 4, lm = lane & 15;
#pragma unroll
  for (int r = 0; r < 8; ++r) {
    int row = tm * 16 + r + 8 * lh;               // D-layout: M = r + 8*half
    size_t rb = (size_t)row * ldc + tn * 16 + lm; // N = lane&15
    C[rb]      = acc0[r];
    C[rb + 16] = acc1[r];
    C[rb + 32] = acc2[r];
    C[rb + 48] = acc3[r];
  }
}

// Adjacency: (Hinc * Hinc^T) > 0, written directly as a byte mask.
// f16 is exact for the 0/1 incidence entries (counts <= 512). Same 4-wide
// N blocking + software pipelining (largest GEMM: 2048x2048x512).
__global__ __launch_bounds__(256) void adj_pk_kernel(
    const _Float16* __restrict__ Apk, const _Float16* __restrict__ Bpk,
    unsigned char* __restrict__ adjb, int tilesM, int tilesN, int kt, int ldc)
{
  int wave = (blockIdx.x * 256 + threadIdx.x) >> 5;
  int lane = threadIdx.x & 31;
  int tn4 = tilesN >> 2;
  if (wave >= tilesM * tn4) return;
  int tm = wave / tn4;
  int tn = (wave % tn4) << 2;
  const _Float16* ap = Apk + ((size_t)tm * kt) * 512 + lane * 16;
  const _Float16* bp = Bpk + ((size_t)tn * kt) * 512 + lane * 16;
  size_t bs = (size_t)kt * 512;
  v8f acc0 = {}, acc1 = {}, acc2 = {}, acc3 = {};

  v16h va  = *(const v16h*)(ap);
  v16h vb0 = *(const v16h*)(bp);
  v16h vb1 = *(const v16h*)(bp + bs);
  v16h vb2 = *(const v16h*)(bp + 2 * bs);
  v16h vb3 = *(const v16h*)(bp + 3 * bs);
  for (int k = 0; k + 1 < kt; ++k) {
    size_t o = (size_t)(k + 1) * 512;
    v16h na  = *(const v16h*)(ap + o);
    v16h nb0 = *(const v16h*)(bp + o);
    v16h nb1 = *(const v16h*)(bp + o + bs);
    v16h nb2 = *(const v16h*)(bp + o + 2 * bs);
    v16h nb3 = *(const v16h*)(bp + o + 3 * bs);
    acc0 = wmma16x16x32(va, vb0, acc0);
    acc1 = wmma16x16x32(va, vb1, acc1);
    acc2 = wmma16x16x32(va, vb2, acc2);
    acc3 = wmma16x16x32(va, vb3, acc3);
    va = na; vb0 = nb0; vb1 = nb1; vb2 = nb2; vb3 = nb3;
  }
  acc0 = wmma16x16x32(va, vb0, acc0);
  acc1 = wmma16x16x32(va, vb1, acc1);
  acc2 = wmma16x16x32(va, vb2, acc2);
  acc3 = wmma16x16x32(va, vb3, acc3);

  int lh = lane >> 4, lm = lane & 15;
#pragma unroll
  for (int r = 0; r < 8; ++r) {
    int row = tm * 16 + r + 8 * lh;
    size_t rb = (size_t)row * ldc + tn * 16 + lm;
    adjb[rb]      = (acc0[r] > 0.5f) ? 1 : 0;
    adjb[rb + 16] = (acc1[r] > 0.5f) ? 1 : 0;
    adjb[rb + 32] = (acc2[r] > 0.5f) ? 1 : 0;
    adjb[rb + 48] = (acc3[r] > 0.5f) ? 1 : 0;
  }
}

// ---------------------------------------------------------------------------
// Flash-style masked multi-head attention. One wave per (head, 16-query tile).
// Head dim = 32 = exactly one WMMA K, so scores cost 1 wmma per 16x16 tile.
// P (probabilities) is re-laid-out into the A operand via an LDS round trip.
// K and V operand loads are hoisted to the chunk top so they overlap the
// softmax VALU work.  Qpk/Kpk: per-head [rows/16][512];
// Vtpk: per-head [2 d-tiles][C/32][512].
// adjb == nullptr -> unmasked.  Output row-major [rows][256].
// ---------------------------------------------------------------------------
__global__ __launch_bounds__(32) void flash_attn_kernel(
    const _Float16* __restrict__ Qpk, const _Float16* __restrict__ Kpk,
    const _Float16* __restrict__ Vtpk, const unsigned char* __restrict__ adjb,
    float* __restrict__ outp, int R, int C)
{
  __shared__ __align__(16) _Float16 lds_p[16 * 32];
  int rtiles = R >> 4;
  int ntile = blockIdx.x % rtiles;
  int head  = blockIdx.x / rtiles;
  int lane = threadIdx.x;
  int lh = lane >> 4, lm = lane & 15;
  int ckt = C >> 5;

  const _Float16* qb = Qpk  + (size_t)head * rtiles * 512;
  const _Float16* kb = Kpk  + (size_t)head * (C >> 4) * 512;
  const _Float16* vb = Vtpk + (size_t)head * 2 * ckt * 512;

  v16h q = *(const v16h*)(qb + (size_t)ntile * 512 + lane * 16); // loop-invariant
  v8f acc0 = {}, acc1 = {};
  float rmax[8], rsum[8];
#pragma unroll
  for (int r = 0; r < 8; ++r) { rmax[r] = -1e30f; rsum[r] = 0.f; }
  const float scale = 0.17677669529663687f;   // 1/sqrt(32)

  for (int m0 = 0; m0 < C; m0 += 32) {
    // issue all global operand loads up-front: K for scores, V for P*V
    v16h k0 = *(const v16h*)(kb + (size_t)(m0 >> 4) * 512 + lane * 16);
    v16h k1 = *(const v16h*)(kb + (size_t)((m0 >> 4) + 1) * 512 + lane * 16);
    v16h v0 = *(const v16h*)(vb + ((size_t)0 * ckt + (m0 >> 5)) * 512 + lane * 16);
    v16h v1 = *(const v16h*)(vb + ((size_t)1 * ckt + (m0 >> 5)) * 512 + lane * 16);
    v8f z = {};
    v8f s0 = wmma16x16x32(q, k0, z);
    v8f s1 = wmma16x16x32(q, k1, z);

    float a0[8], a1[8];
#pragma unroll
    for (int r = 0; r < 8; ++r) {
      float x0 = s0[r] * scale, x1 = s1[r] * scale;
      if (adjb) {
        int row = ntile * 16 + r + 8 * lh;
        x0 = adjb[(size_t)row * C + m0 + lm]      ? x0 : -1e30f;
        x1 = adjb[(size_t)row * C + m0 + 16 + lm] ? x1 : -1e30f;
      }
      a0[r] = x0; a1[r] = x1;
    }
    // row max: columns of one row live across the 16 lanes of a half-wave
    float tmax[8];
#pragma unroll
    for (int r = 0; r < 8; ++r) tmax[r] = fmaxf(a0[r], a1[r]);
#pragma unroll
    for (int off = 1; off < 16; off <<= 1) {
#pragma unroll
      for (int r = 0; r < 8; ++r)
        tmax[r] = fmaxf(tmax[r], __shfl_xor(tmax[r], off, 32));
    }
    float psum[8];
#pragma unroll
    for (int r = 0; r < 8; ++r) {
      float mnew = fmaxf(rmax[r], tmax[r]);
      float corr = __expf(rmax[r] - mnew);
      float p0 = __expf(a0[r] - mnew);
      float p1 = __expf(a1[r] - mnew);
      acc0[r] = acc0[r] * corr;
      acc1[r] = acc1[r] * corr;
      rsum[r] = rsum[r] * corr;
      rmax[r] = mnew;
      psum[r] = p0 + p1;
      int prow = r + 8 * lh;                       // full 0..15 across halves
      lds_p[prow * 32 + lm]      = (_Float16)p0;
      lds_p[prow * 32 + 16 + lm] = (_Float16)p1;
    }
#pragma unroll
    for (int off = 1; off < 16; off <<= 1) {
#pragma unroll
      for (int r = 0; r < 8; ++r)
        psum[r] += __shfl_xor(psum[r], off, 32);
    }
#pragma unroll
    for (int r = 0; r < 8; ++r) rsum[r] += psum[r];

    // re-layout P (row-major 16x32 in LDS) into the WMMA A operand
    asm volatile("s_wait_dscnt 0" ::: "memory");   // DS RAW, same wave
    v8h plo = *(const v8h*)(&lds_p[lm * 32 + 8 * lh]);
    v8h phi = *(const v8h*)(&lds_p[lm * 32 + 16 + 8 * lh]);
    v16h pa = __builtin_shufflevector(plo, phi, 0, 1, 2, 3, 4, 5, 6, 7,
                                      8, 9, 10, 11, 12, 13, 14, 15);
    acc0 = wmma16x16x32(pa, v0, acc0);
    acc1 = wmma16x16x32(pa, v1, acc1);
  }
#pragma unroll
  for (int r = 0; r < 8; ++r) {
    int n = ntile * 16 + r + 8 * lh;
    float inv = 1.0f / rsum[r];
    outp[(size_t)n * DD + head * HDIM + lm]      = acc0[r] * inv;
    outp[(size_t)n * DD + head * HDIM + 16 + lm] = acc1[r] * inv;
  }
}

// ---------------------------------------------------------------------------
// Sigmoid gates: g_n, g_e (mean over heads), spec gate, normalized 3-way split.
// ---------------------------------------------------------------------------
__global__ __launch_bounds__(256) void gates_kernel(
    const float* __restrict__ Q,
    const float* __restrict__ gn_w, const float* __restrict__ gn_b,
    const float* __restrict__ ge_w, const float* __restrict__ ge_b,
    const float* __restrict__ gs_w, const float* __restrict__ gs_b,
    float* __restrict__ gate4)
{
  int n = blockIdx.x * 256 + threadIdx.x;
  if (n >= NN) return;
  float an[NHEADS] = {}, ae[NHEADS] = {}, as_[NHEADS] = {};
  const float* qr = Q + (size_t)n * DD;
  for (int i = 0; i < DD; ++i) {
    float qv = qr[i];
#pragma unroll
    for (int h = 0; h < NHEADS; ++h) {
      an[h]  += qv * gn_w[h * DD + i];
      ae[h]  += qv * ge_w[h * DD + i];
      as_[h] += qv * gs_w[h * DD + i];
    }
  }
  float gn = 0.f, ge = 0.f, sg = 0.f;
#pragma unroll
  for (int h = 0; h < NHEADS; ++h) {
    gn += 1.0f / (1.0f + __expf(-(an[h]  + gn_b[h])));
    ge += 1.0f / (1.0f + __expf(-(ae[h]  + ge_b[h])));
    sg += 1.0f / (1.0f + __expf(-(as_[h] + gs_b[h])));
  }
  gn *= 0.125f; ge *= 0.125f; sg *= 0.125f;
  float gs = fmaxf(1.0f - gn - ge, 0.0f);
  float tot = gn + ge + gs + 1e-8f;
  gate4[n * 4 + 0] = gn / tot;
  gate4[n * 4 + 1] = ge / tot;
  gate4[n * 4 + 2] = gs / tot;
  gate4[n * 4 + 3] = sg;
}

__global__ __launch_bounds__(256) void combine_kernel(
    const float* __restrict__ on_, const float* __restrict__ oe_,
    const float* __restrict__ sf_, const float* __restrict__ gate4,
    float* __restrict__ comb)
{
  int idx = blockIdx.x * 256 + threadIdx.x;
  if (idx >= NN * DD) return;
  int n = idx >> 8;  // DD == 256
  const float* g = gate4 + n * 4;
  comb[idx] = g[0] * on_[idx] + g[1] * oe_[idx] + g[2] * (g[3] * sf_[idx]);
}

// bias + residual + LayerNorm over D=256; one 256-thread block per row
__global__ __launch_bounds__(256) void bias_res_ln_kernel(
    const float* __restrict__ lin, const float* __restrict__ bias,
    const float* __restrict__ x, const float* __restrict__ gamma,
    const float* __restrict__ beta, float* __restrict__ out)
{
  __shared__ float red[DD];
  int n = blockIdx.x, d = threadIdx.x;
  float v = lin[(size_t)n * DD + d] + bias[d] + x[(size_t)n * DD + d];
  red[d] = v; __syncthreads();
  for (int s = 128; s > 0; s >>= 1) { if (d < s) red[d] += red[d + s]; __syncthreads(); }
  float mu = red[0] * (1.0f / DD);
  __syncthreads();
  float c = v - mu;
  red[d] = c * c; __syncthreads();
  for (int s = 128; s > 0; s >>= 1) { if (d < s) red[d] += red[d + s]; __syncthreads(); }
  float var = red[0] * (1.0f / DD);
  out[(size_t)n * DD + d] = gamma[d] * c * rsqrtf(var + 1e-5f) + beta[d];
}

// ---------------------------------------------------------------------------
extern "C" void kernel_launch(void* const* d_in, const int* in_sizes, int n_in,
                              void* d_out, int out_size, void* d_ws, size_t ws_size,
                              hipStream_t stream)
{
  (void)in_sizes; (void)n_in; (void)out_size; (void)ws_size;
  const float* x     = (const float*)d_in[0];
  const float* Hinc  = (const float*)d_in[1];
  const float* U_r   = (const float*)d_in[2];
  const float* Wq    = (const float*)d_in[3];
  const float* Wk    = (const float*)d_in[4];
  const float* Wv    = (const float*)d_in[5];
  const float* Wspec = (const float*)d_in[6];
  const float* gn_w  = (const float*)d_in[7];
  const float* gn_b  = (const float*)d_in[8];
  const float* ge_w  = (const float*)d_in[9];
  const float* ge_b  = (const float*)d_in[10];
  const float* gs_w  = (const float*)d_in[11];
  const float* gs_b  = (const float*)d_in[12];
  const float* Woutw = (const float*)d_in[13];
  const float* Woutb = (const float*)d_in[14];
  const float* gamma = (const float*)d_in[15];
  const float* beta  = (const float*)d_in[16];
  float* out = (float*)d_out;

  char* base = (char*)d_ws;
  size_t off = 0;
  auto alloc = [&](size_t bytes) -> void* {
    void* p = base + off;
    off += (bytes + 255) & ~(size_t)255;
    return p;
  };

  // f16 packed operands
  _Float16* xpk      = (_Float16*)alloc((size_t)NN * DD * 2);
  _Float16* wqpk     = (_Float16*)alloc((size_t)DD * DD * 2);
  _Float16* wkpk     = (_Float16*)alloc((size_t)DD * DD * 2);
  _Float16* wvpk     = (_Float16*)alloc((size_t)DD * DD * 2);
  _Float16* woutpk   = (_Float16*)alloc((size_t)DD * DD * 2);
  _Float16* wspk     = (_Float16*)alloc((size_t)DD * RANK_ * 2);
  _Float16* urpk     = (_Float16*)alloc((size_t)NN * RANK_ * 2);
  _Float16* hpk      = (_Float16*)alloc((size_t)NN * MM * 2);
  _Float16* htpk     = (_Float16*)alloc((size_t)MM * NN * 2);
  _Float16* xtpk     = (_Float16*)alloc((size_t)DD * NN * 2);
  _Float16* qhpk     = (_Float16*)alloc((size_t)NN * DD * 2);   // 8 heads x [N,32]
  _Float16* khpk     = (_Float16*)alloc((size_t)NN * DD * 2);
  _Float16* vtpk     = (_Float16*)alloc((size_t)NN * DD * 2);   // 8 heads x [32,N]
  _Float16* hepk     = (_Float16*)alloc((size_t)MM * DD * 2);
  _Float16* qepk     = (_Float16*)alloc((size_t)MM * DD * 2);
  _Float16* kepk     = (_Float16*)alloc((size_t)MM * DD * 2);
  _Float16* vtepk    = (_Float16*)alloc((size_t)MM * DD * 2);
  _Float16* heoutTpk = (_Float16*)alloc((size_t)DD * MM * 2);
  _Float16* combpk   = (_Float16*)alloc((size_t)NN * DD * 2);
  // f32 intermediates
  float* Q        = (float*)alloc((size_t)NN * DD * 4);
  float* Km       = (float*)alloc((size_t)NN * DD * 4);
  float* V        = (float*)alloc((size_t)NN * DD * 4);
  float* out_node = (float*)alloc((size_t)NN * DD * 4);
  float* out_edge = (float*)alloc((size_t)NN * DD * 4);
  float* spec     = (float*)alloc((size_t)NN * DD * 4);
  float* h_e      = (float*)alloc((size_t)MM * DD * 4);
  float* Qe       = (float*)alloc((size_t)MM * DD * 4);
  float* Ke       = (float*)alloc((size_t)MM * DD * 4);
  float* Ve       = (float*)alloc((size_t)MM * DD * 4);
  float* h_e_out  = (float*)alloc((size_t)MM * DD * 4);
  float* gate4    = (float*)alloc((size_t)NN * 4 * 4);
  float* comb     = (float*)alloc((size_t)NN * DD * 4);
  float* out_lin  = (float*)alloc((size_t)NN * DD * 4);
  unsigned char* adjb = (unsigned char*)alloc((size_t)NN * NN);

  auto PACK = [&](const float* s, _Float16* dpk, int R, int K, int lda, int tr) {
    int total = R * K;
    pack_f16_kernel<<<(total + 255) / 256, 256, 0, stream>>>(s, dpk, R, K, lda, tr);
  };
  auto GEMM = [&](const _Float16* A, const _Float16* B, float* Cm,
                  int Mr, int Nr, int Kd, int ldc) {
    int tm = Mr / 16, tn = Nr / 16, kt = Kd / 32;
    int waves = tm * (tn / 4);            // 16x64 strip per wave
    gemm_pk_kernel<<<(waves + 7) / 8, 256, 0, stream>>>(A, B, Cm, tm, tn, kt, ldc);
  };

  // --- pack static operands ---
  PACK(x,     xpk,    NN, DD,    DD,    0);
  PACK(Wq,    wqpk,   DD, DD,    DD,    0);
  PACK(Wk,    wkpk,   DD, DD,    DD,    0);
  PACK(Wv,    wvpk,   DD, DD,    DD,    0);
  PACK(Woutw, woutpk, DD, DD,    DD,    0);
  PACK(Wspec, wspk,   DD, RANK_, RANK_, 0);
  PACK(U_r,   urpk,   NN, RANK_, RANK_, 0);
  PACK(Hinc,  hpk,    NN, MM,    MM,    0);
  PACK(Hinc,  htpk,   MM, NN,    MM,    1);   // Hinc^T
  PACK(x,     xtpk,   DD, NN,    DD,    1);   // x^T

  // --- projections Q/K/V = x @ W^T ---
  GEMM(xpk, wqpk, Q,  NN, DD, DD, DD);
  GEMM(xpk, wkpk, Km, NN, DD, DD, DD);
  GEMM(xpk, wvpk, V,  NN, DD, DD, DD);

  // --- adjacency mask (Hinc Hinc^T > 0) via WMMA, byte output ---
  {
    int tm = NN / 16, tn = NN / 16, kt = MM / 32;
    int waves = tm * (tn / 4);
    adj_pk_kernel<<<(waves + 7) / 8, 256, 0, stream>>>(hpk, hpk, adjb, tm, tn, kt, NN);
  }

  // --- per-head repack for node attention ---
  for (int h = 0; h < NHEADS; ++h) {
    PACK(Q  + h * HDIM, qhpk + (size_t)h * NN * HDIM, NN,   HDIM, DD, 0);
    PACK(Km + h * HDIM, khpk + (size_t)h * NN * HDIM, NN,   HDIM, DD, 0);
    PACK(V  + h * HDIM, vtpk + (size_t)h * HDIM * NN, HDIM, NN,   DD, 1); // V^T
  }
  gates_kernel<<<(NN + 255) / 256, 256, 0, stream>>>(Q, gn_w, gn_b, ge_w, ge_b,
                                                     gs_w, gs_b, gate4);
  // --- node-level masked flash attention ---
  flash_attn_kernel<<<(NN / 16) * NHEADS, 32, 0, stream>>>(
      qhpk, khpk, vtpk, adjb, out_node, NN, NN);

  // --- hyperedge branch: h_e = Hinc^T x, projections, unmasked attention ---
  GEMM(htpk, xtpk, h_e, MM, DD, NN, DD);
  PACK(h_e, hepk, MM, DD, DD, 0);
  GEMM(hepk, wqpk, Qe, MM, DD, DD, DD);
  GEMM(hepk, wkpk, Ke, MM, DD, DD, DD);
  GEMM(hepk, wvpk, Ve, MM, DD, DD, DD);
  for (int h = 0; h < NHEADS; ++h) {
    PACK(Qe + h * HDIM, qepk  + (size_t)h * MM * HDIM, MM,   HDIM, DD, 0);
    PACK(Ke + h * HDIM, kepk  + (size_t)h * MM * HDIM, MM,   HDIM, DD, 0);
    PACK(Ve + h * HDIM, vtepk + (size_t)h * HDIM * MM, HDIM, MM,   DD, 1);
  }
  flash_attn_kernel<<<(MM / 16) * NHEADS, 32, 0, stream>>>(
      qepk, kepk, vtepk, nullptr, h_e_out, MM, MM);
  PACK(h_e_out, heoutTpk, DD, MM, DD, 1);          // h_e_out^T
  GEMM(hpk, heoutTpk, out_edge, NN, DD, MM, DD);   // Hinc @ h_e_out

  // --- spectral branch ---
  GEMM(urpk, wspk, spec, NN, DD, RANK_, DD);

  // --- gated combine, output projection, residual + LayerNorm ---
  combine_kernel<<<(NN * DD) / 256, 256, 0, stream>>>(out_node, out_edge, spec,
                                                      gate4, comb);
  PACK(comb, combpk, NN, DD, DD, 0);
  GEMM(combpk, woutpk, out_lin, NN, DD, DD, DD);
  bias_res_ln_kernel<<<NN, DD, 0, stream>>>(out_lin, Woutb, x, gamma, beta, out);
}
